// TSDiffusion_17970143166670
// MI455X (gfx1250) — matrix operand order repaired and checked by reading, not compiled, not run on hardware
//
#include <hip/hip_runtime.h>

#define B_  16
#define T_  1024
#define C_  16
#define D_  256
#define H_  4
#define HD_ 64
#define BT_ (B_*T_)

typedef __attribute__((ext_vector_type(16))) _Float16 v16h;
typedef __attribute__((ext_vector_type(8)))  float    v8f;

__device__ __forceinline__ v8f wmma32(v16h a, v16h b, v8f c){
  return __builtin_amdgcn_wmma_f32_16x16x32_f16(false, a, false, b, (short)0, c, false, false);
}

__device__ __forceinline__ float tanh_fast(float x){
  float xc = fminf(fmaxf(x, -10.f), 10.f);
  float e = __expf(2.f*xc);
  return __fdividef(e-1.f, e+1.f);
}
__device__ __forceinline__ float sigmoid_fast(float x){
  return __fdividef(1.f, 1.f + __expf(-x));
}

// A fragment (16x32 f16, MxK): lane l -> M=l%16, K=i+16*(l/16)
__device__ __forceinline__ v16h load_a_frag(const _Float16* A, int lda, int k0){
  int lane = threadIdx.x & 31;
  const _Float16* p = A + (size_t)(lane & 15)*lda + k0 + 16*(lane>>4);
  v16h a;
#pragma unroll
  for (int i=0;i<8;++i) a[i] = p[i];
#pragma unroll
  for (int i=0;i<8;++i) a[i+8] = p[i+8];
  return a;
}

// B fragment from fragment-major (pre-shuffled) weights: one 32B contiguous load per lane
__device__ __forceinline__ v16h load_b_frag_pre(const _Float16* Wf, int ntn, int kt, int nt){
  int lane = threadIdx.x & 31;
  return *(const v16h*)(Wf + ((((size_t)kt*ntn + nt)*32) + lane)*16);
}

// B fragment where source is row-major NxK (i.e. B[k][n] = S[n][k]) -- contiguous per lane
__device__ __forceinline__ v16h load_b_fragT(const _Float16* Sm, int lds_, int n0, int k0){
  int lane = threadIdx.x & 31;
  const _Float16* p = Sm + (size_t)(n0 + (lane & 15))*lds_ + k0 + 16*(lane>>4);
  v16h b;
#pragma unroll
  for (int i=0;i<16;++i) b[i] = p[i];
  return b;
}

// two independent 16x16 tiles sharing one A stream (K=256) -> 2x WMMA ILP.
// The entry memory-clobber stops LICM from hoisting the (loop-invariant) weight
// fragments out of the 1024-step scan loop, which was spilling ~1024 VGPRs to scratch.
__device__ __forceinline__ void wave_gemm_pair(const _Float16* Asrc, int lda,
    const _Float16* Wf, int ntn, int nt0, int nt1,
    const float* bias, v8f& c0, v8f& c1)
{
  asm volatile("" ::: "memory");
  int n16 = threadIdx.x & 15;
#pragma unroll
  for (int i=0;i<8;++i){ c0[i] = bias[nt0*16+n16]; c1[i] = bias[nt1*16+n16]; }
#pragma unroll
  for (int kt=0; kt<8; ++kt){
    v16h a = load_a_frag(Asrc, lda, kt*32);
    c0 = wmma32(a, load_b_frag_pre(Wf, ntn, kt, nt0), c0);
    c1 = wmma32(a, load_b_frag_pre(Wf, ntn, kt, nt1), c1);
  }
}

// ---------------- weight conversion + fragment-major shuffle ----------------
__global__ void k_cvt_frag(const float* __restrict__ W, _Float16* __restrict__ out, int K, int N){
  int idx = blockIdx.x*256 + threadIdx.x;
  int total = (K>>5)*(N>>4)*32;
  if (idx >= total) return;
  int lane = idx & 31;
  int tile = idx >> 5;
  int ntn = N>>4;
  int kt = tile / ntn, nt = tile - kt*ntn;
  int n  = nt*16 + (lane & 15);
  int kb = kt*32 + 16*(lane>>4);
  _Float16* dst = out + (size_t)idx*16;
#pragma unroll
  for (int i=0;i<16;++i) dst[i] = (_Float16)W[(size_t)(kb+i)*N + n];
}

// ---------------- te / FiLM (tiny, scalar) ----------------
__global__ __launch_bounds__(256) void k_te_film(
    const int* __restrict__ t, const float* __restrict__ te_freqs,
    const float* __restrict__ W_temb, const float* __restrict__ b_temb,
    const float* __restrict__ W_film, const float* __restrict__ b_film,
    float* __restrict__ te, float* __restrict__ gb)
{
  __shared__ float s_pre[B_*D_];
  __shared__ float s_silu[B_*D_];
  int tid = threadIdx.x;
  for (int idx=tid; idx<B_*D_; idx+=256){
    int b = idx>>8, j = idx&255;
    float tf = (float)t[b];
    s_pre[idx] = (j<128) ? sinf(tf*te_freqs[j]) : cosf(tf*te_freqs[j-128]);
  }
  __syncthreads();
  for (int idx=tid; idx<B_*D_; idx+=256){
    int b = idx>>8, d = idx&255;
    float acc = b_temb[d];
    for (int k=0;k<D_;++k) acc += s_pre[b*D_+k]*W_temb[k*D_+d];
    te[idx] = acc;
    s_silu[idx] = acc/(1.f+expf(-acc));
  }
  __syncthreads();
  for (int idx=tid; idx<B_*2*D_; idx+=256){
    int b = idx>>9, j = idx&511;
    float acc = b_film[j];
    for (int k=0;k<D_;++k) acc += s_silu[b*D_+k]*W_film[k*2*D_+j];
    gb[idx] = acc;
  }
}

// ---------------- encode + diffusion + time embeddings, fused ----------------
__global__ __launch_bounds__(256) void k_embed(
    const float* __restrict__ x, const int* __restrict__ t, const float* __restrict__ tsg,
    const float* __restrict__ mask, const float* __restrict__ noise,
    const float* __restrict__ alpha_bar, const float* __restrict__ W_enc, const float* __restrict__ b_enc,
    const float* __restrict__ th_a, const float* __restrict__ th_b, const float* __restrict__ th_freqs,
    const float* __restrict__ te, const float* __restrict__ gb,
    _Float16* __restrict__ h16)
{
  __shared__ float sx[C_], sm[C_];
  int pos = blockIdx.x;
  int b = pos >> 10;
  int d = threadIdx.x;
  if (d < C_) sx[d] = x[(size_t)pos*C_ + d];
  else if (d < 2*C_) sm[d-C_] = mask[(size_t)pos*C_ + (d-C_)];
  __syncthreads();
  float acc = b_enc[d];
  for (int k=0;k<C_;++k) acc += sx[k]*W_enc[k*D_+d] + sm[k]*W_enc[(C_+k)*D_+d];
  float z = fmaxf(acc, 0.f);
  float ab = alpha_bar[t[b]];
  float hv = sqrtf(ab)*z + sqrtf(1.f-ab)*noise[(size_t)pos*D_+d];
  float ts = tsg[pos];
  float tmv;
  if (d & 1){ int j = d>>1; tmv = ts*th_a[j] + th_b[j]; }
  else { int j = d>>1; float ph = ts*th_freqs[j>>1]; tmv = (j&1) ? cosf(ph) : sinf(ph); }
  float g = gb[b*2*D_ + d], be = gb[b*2*D_ + D_ + d];
  tmv = (1.f+g)*tmv + be;
  hv += te[b*D_+d] + tmv;
  h16[(size_t)pos*D_ + d] = (_Float16)hv;
}

// ---------------- persistent RK4 + GRU scan (1 workgroup, state in LDS) ----------------
__global__ __launch_bounds__(256) void k_scan(
    const _Float16* __restrict__ h16, const float* __restrict__ tsg,
    const _Float16* __restrict__ Wih, const _Float16* __restrict__ Whh,   // fragment-major
    const float* __restrict__ bih, const float* __restrict__ bhh,
    const _Float16* __restrict__ oW1, const float* __restrict__ ob1,      // fragment-major
    const _Float16* __restrict__ oW2, const float* __restrict__ ob2,      // fragment-major
    float* __restrict__ Hseq)
{
  __shared__ __align__(16) char smem[65536];
  float*    s_hh   = (float*)smem;                 // 16x256 f32   [0,16384)
  _Float16* s_cand = (_Float16*)(smem + 16384);    // 16x256 f16   [16384,24576)  A-source
  float*    s_acc  = (float*)(smem + 24576);       // 16x256 f32   (ODE phase)
  _Float16* s_mid  = (_Float16*)(smem + 40960);    // 16x256 f16   (ODE phase)
  _Float16* s_gi   = (_Float16*)(smem + 24576);    // 16x768 f16   (GRU phase, aliases)
  _Float16* s_r    = (_Float16*)(smem + 49152);    // 16x256 f16
  _Float16* s_z    = (_Float16*)(smem + 57344);    // 16x256 f16

  int tid = threadIdx.x, wave = tid>>5, lane = tid&31;
  int n16 = lane & 15, hi = lane >> 4;
  for (int i=tid;i<16*D_;i+=256){ s_hh[i]=0.f; s_cand[i]=(_Float16)0.f; }
  __syncthreads();

#pragma unroll 1
  for (int tt=0; tt<T_; ++tt){
    asm volatile("" ::: "memory");
    for (int i=tid;i<16*D_;i+=256) s_acc[i]=0.f;
    __syncthreads();

#pragma unroll 1
    for (int stage=0; stage<4; ++stage){
      // mid = tanh(cand @ ode_W1 + b1)
      {
        v8f c0, c1;
        wave_gemm_pair(s_cand, D_, oW1, 16, wave, wave+8, ob1, c0, c1);
        int mb = 8*hi, na = wave*16+n16, nb = (wave+8)*16+n16;
#pragma unroll
        for (int i=0;i<8;++i){
          s_mid[(mb+i)*D_ + na] = (_Float16)tanh_fast(c0[i]);
          s_mid[(mb+i)*D_ + nb] = (_Float16)tanh_fast(c1[i]);
        }
      }
      __syncthreads();
      // f = mid @ ode_W2 + b2 ; acc += w*f ; cand = hh + alpha*dt*f
      {
        float wc    = (stage==0 || stage==3) ? 1.f : 2.f;
        float alpha = (stage<2) ? 0.5f : 1.f;
        v8f c0, c1;
        wave_gemm_pair(s_mid, D_, oW2, 16, wave, wave+8, ob2, c0, c1);
        int mb = 8*hi, na = wave*16+n16, nb = (wave+8)*16+n16;
#pragma unroll
        for (int i=0;i<8;++i){
          int m = mb+i;
          float dtm = (tt==0) ? 0.f : (tsg[m*T_+tt]-tsg[m*T_+tt-1]);
          s_acc[m*D_+na] += wc*c0[i];
          s_acc[m*D_+nb] += wc*c1[i];
          if (stage<3){
            s_cand[m*D_+na] = (_Float16)(s_hh[m*D_+na] + alpha*dtm*c0[i]);
            s_cand[m*D_+nb] = (_Float16)(s_hh[m*D_+nb] + alpha*dtm*c1[i]);
          }
        }
      }
      __syncthreads();
    }
    // RK4 combine
    for (int idx=tid; idx<16*D_; idx+=256){
      int m = idx>>8;
      float dtm = (tt==0) ? 0.f : (tsg[m*T_+tt]-tsg[m*T_+tt-1]);
      float v = s_hh[idx] + dtm*(1.f/6.f)*s_acc[idx];
      s_hh[idx]=v; s_cand[idx]=(_Float16)v;
    }
    __syncthreads();
    // gi = x_t @ Wih + bih   (A streamed from global h16, row stride T*D)
    const _Float16* Ax = h16 + (size_t)tt*D_;
#pragma unroll 1
    for (int p=0;p<3;++p){
      int nt0 = wave + p*16, nt1 = nt0 + 8;
      v8f c0, c1;
      wave_gemm_pair(Ax, T_*D_, Wih, 48, nt0, nt1, bih, c0, c1);
      int mb = 8*hi, na = nt0*16+n16, nb = nt1*16+n16;
#pragma unroll
      for (int i=0;i<8;++i){
        s_gi[(mb+i)*3*D_ + na] = (_Float16)c0[i];
        s_gi[(mb+i)*3*D_ + nb] = (_Float16)c1[i];
      }
    }
    __syncthreads();
    // gh pass 1: r and z gates (cols 0..511)
#pragma unroll 1
    for (int p=0;p<2;++p){
      int nt0 = wave + p*16, nt1 = nt0 + 8;
      v8f c0, c1;
      wave_gemm_pair(s_cand, D_, Whh, 48, nt0, nt1, bhh, c0, c1);
      int mb = 8*hi, na = nt0*16+n16, nb = nt1*16+n16;
#pragma unroll
      for (int i=0;i<8;++i){
        int m = mb+i;
        float sgA = sigmoid_fast(c0[i] + (float)s_gi[m*3*D_+na]);
        float sgB = sigmoid_fast(c1[i] + (float)s_gi[m*3*D_+nb]);
        if (na < D_) s_r[m*D_+na] = (_Float16)sgA; else s_z[m*D_+na-D_] = (_Float16)sgA;
        if (nb < D_) s_r[m*D_+nb] = (_Float16)sgB; else s_z[m*D_+nb-D_] = (_Float16)sgB;
      }
    }
    __syncthreads();
    // gh pass 2: candidate gate + state update (cols 512..767)
    {
      v8f c0, c1;
      wave_gemm_pair(s_cand, D_, Whh, 48, 32+wave, 40+wave, bhh, c0, c1);
      __syncthreads();           // all s_cand reads done before commit
      int na = (32+wave)*16+n16, nb = (40+wave)*16+n16;
      int jA = na-2*D_, jB = nb-2*D_;
      int mb = 8*hi;
#pragma unroll
      for (int i=0;i<8;++i){
        int m = mb+i;
        float nvA = tanh_fast((float)s_gi[m*3*D_+na] + (float)s_r[m*D_+jA]*c0[i]);
        float nvB = tanh_fast((float)s_gi[m*3*D_+nb] + (float)s_r[m*D_+jB]*c1[i]);
        float zgA = (float)s_z[m*D_+jA];
        float zgB = (float)s_z[m*D_+jB];
        float hA = (1.f-zgA)*nvA + zgA*s_hh[m*D_+jA];
        float hB = (1.f-zgB)*nvB + zgB*s_hh[m*D_+jB];
        s_hh[m*D_+jA] = hA;  s_cand[m*D_+jA] = (_Float16)hA;
        s_hh[m*D_+jB] = hB;  s_cand[m*D_+jB] = (_Float16)hB;
        Hseq[((size_t)(m*T_+tt))*D_ + jA] = hA;
        Hseq[((size_t)(m*T_+tt))*D_ + jB] = hB;
      }
    }
    __syncthreads();
  }
}

// ---------------- block-tiled WMMA GEMM: 16 x 256 output per block ----------------
// A tile staged in LDS via GLOBAL_LOAD_ASYNC_TO_LDS_B128 (ASYNCcnt); each wave owns 2 N-tiles
__global__ __launch_bounds__(256) void k_gemm(
    const _Float16* __restrict__ A, const _Float16* __restrict__ Bf,
    const float* __restrict__ bias, const float* __restrict__ resid,
    float* __restrict__ outF, _Float16* __restrict__ outH,
    int M, int N, int K, int relu)
{
  extern __shared__ __align__(16) _Float16 s_a[];   // 16 x K
  int tid = threadIdx.x;
  int nblk = N >> 8;                  // column blocks of 256
  int bm = blockIdx.x / nblk;
  int bn = blockIdx.x - bm*nblk;
  // async-stage the 16 x K A tile into LDS (one b128 per lane per iteration)
  int cpr = K >> 3;                   // 8-half chunks per row
  for (int idx=tid; idx<16*cpr; idx+=256){
    int row = idx / cpr, co = (idx - row*cpr)*8;
    unsigned ldsa = (unsigned)(unsigned long long)&s_a[row*K + co];
    unsigned long long ga = (unsigned long long)(A + ((size_t)(bm*16+row))*K + co);
    asm volatile("global_load_async_to_lds_b128 %0, %1, off" :: "v"(ldsa), "v"(ga) : "memory");
  }
  asm volatile("s_wait_asynccnt 0x0" ::: "memory");
  __syncthreads();
  int wave = tid>>5, lane = tid&31;
  int tn0 = bn*16 + wave, tn1 = tn0 + 8;
  int ntn = N >> 4;
  int n0 = tn0*16 + (lane & 15), n1 = tn1*16 + (lane & 15);
  int mb = bm*16 + 8*(lane >> 4);
  v8f c0, c1;
#pragma unroll
  for (int i=0;i<8;++i){
    c0[i] = bias[n0] + (resid ? resid[(size_t)(mb+i)*N + n0] : 0.f);
    c1[i] = bias[n1] + (resid ? resid[(size_t)(mb+i)*N + n1] : 0.f);
  }
  int ktiles = K >> 5;
  for (int kt=0; kt<ktiles; ++kt){
    if (kt+1 < ktiles)
      __builtin_prefetch(Bf + ((((size_t)(kt+1)*ntn + tn0)*32))*16, 0, 1);
    v16h a = load_a_frag(s_a, K, kt*32);
    c0 = wmma32(a, load_b_frag_pre(Bf, ntn, kt, tn0), c0);
    c1 = wmma32(a, load_b_frag_pre(Bf, ntn, kt, tn1), c1);
  }
  if (relu){
#pragma unroll
    for (int i=0;i<8;++i){ c0[i] = fmaxf(c0[i], 0.f); c1[i] = fmaxf(c1[i], 0.f); }
  }
#pragma unroll
  for (int i=0;i<8;++i){
    size_t iA = (size_t)(mb+i)*N + n0;
    size_t iB = (size_t)(mb+i)*N + n1;
    if (outF){ outF[iA] = c0[i]; outF[iB] = c1[i]; }
    if (outH){ outH[iA] = (_Float16)c0[i]; outH[iB] = (_Float16)c1[i]; }
  }
}

// ---------------- wave-per-row LayerNorm (D=256) ----------------
__global__ __launch_bounds__(256) void k_ln(
    const float* __restrict__ in, const float* __restrict__ w, const float* __restrict__ bv,
    _Float16* __restrict__ outH, float* __restrict__ outF, const float* __restrict__ addsrc)
{
  int row = blockIdx.x*8 + (threadIdx.x>>5);
  int lane = threadIdx.x & 31;
  const float* p = in + (size_t)row*D_ + lane*8;
  float xv[8]; float s=0.f, s2=0.f;
#pragma unroll
  for (int i=0;i<8;++i){ xv[i]=p[i]; s+=xv[i]; s2+=xv[i]*xv[i]; }
#pragma unroll
  for (int m=1;m<32;m<<=1){ s += __shfl_xor(s,m,32); s2 += __shfl_xor(s2,m,32); }
  float mean = s*(1.f/D_);
  float var  = s2*(1.f/D_) - mean*mean;
  float rstd = rsqrtf(var + 1e-5f);
#pragma unroll
  for (int i=0;i<8;++i){
    int d = lane*8+i;
    float v = (xv[i]-mean)*rstd*w[d] + bv[d];
    size_t idx = (size_t)row*D_ + d;
    if (outH) outH[idx] = (_Float16)v;
    if (outF) outF[idx] = v + (addsrc ? addsrc[idx] : 0.f);
  }
}

// ---------------- V transpose: VT[b][h][d][t] = V[b][t][h][d] ----------------
__global__ __launch_bounds__(256) void k_vt(const _Float16* __restrict__ qkv, _Float16* __restrict__ VT)
{
  int pos = blockIdx.x;           // b*T + t
  int b = pos >> 10, t = pos & 1023;
  int d0 = threadIdx.x;           // h*64 + d
  int hh = d0 >> 6, d = d0 & 63;
  VT[(((size_t)(b*H_ + hh))*HD_ + d)*T_ + t] = qkv[(size_t)pos*768 + 2*D_ + d0];
}

// ---------------- flash-style causal attention, 1 wave per (b, head, 16-query tile) ----------------
__global__ __launch_bounds__(32) void k_attn(const _Float16* __restrict__ qkv,
                                             const _Float16* __restrict__ VT,
                                             _Float16* __restrict__ o16)
{
  __shared__ _Float16 s_p[16*32];
  int bid = blockIdx.x;
  int qt = bid & 63, hh = (bid>>6)&3, b = bid>>8;
  int lane = threadIdx.x & 31;
  const int ld = 3*D_;
  const _Float16* Qb  = qkv + ((size_t)(b*T_ + qt*16))*ld + hh*HD_;
  const _Float16* Kb  = qkv + ((size_t)b*T_)*ld + D_ + hh*HD_;
  const _Float16* VTb = VT  + ((size_t)(b*H_ + hh))*HD_*T_;
  v16h aq0 = load_a_frag(Qb, ld, 0);
  v16h aq1 = load_a_frag(Qb, ld, 32);
  v8f o0={},o1={},o2={},o3={};
  float mr[8], lr[8];
#pragma unroll
  for (int i=0;i<8;++i){ mr[i]=-1e30f; lr[i]=0.f; }
  int n = lane & 15, hi = lane >> 4;
  int qmax = qt*16 + 15;
  for (int j0=0; j0<=qmax; j0+=32){
    v8f s0={}, s1={};
    s0 = wmma32(aq0, load_b_fragT(Kb, ld, j0, 0),  s0);
    s0 = wmma32(aq1, load_b_fragT(Kb, ld, j0, 32), s0);
    bool v1 = (j0+16) <= qmax;
    if (v1){
      s1 = wmma32(aq0, load_b_fragT(Kb, ld, j0+16, 0),  s1);
      s1 = wmma32(aq1, load_b_fragT(Kb, ld, j0+16, 32), s1);
    }
#pragma unroll
    for (int i=0;i<8;++i){
      int M = i + 8*hi; int qpos = qt*16 + M;
      float a0 = (j0+n <= qpos) ? s0[i]*0.125f : -1e30f;
      float a1 = (v1 && (j0+16+n <= qpos)) ? s1[i]*0.125f : -1e30f;
      float tmax = fmaxf(a0, a1);
#pragma unroll
      for (int m=1;m<16;m<<=1) tmax = fmaxf(tmax, __shfl_xor(tmax,m,32));
      float newm = fmaxf(mr[i], tmax);
      float scl = __expf(mr[i]-newm);
      float p0 = __expf(a0-newm);
      float p1 = __expf(a1-newm);
      float rs = p0+p1;
#pragma unroll
      for (int m=1;m<16;m<<=1) rs += __shfl_xor(rs,m,32);
      lr[i] = lr[i]*scl + rs;
      mr[i] = newm;
      o0[i]*=scl; o1[i]*=scl; o2[i]*=scl; o3[i]*=scl;
      s_p[M*32 + n]      = (_Float16)p0;
      s_p[M*32 + 16 + n] = (_Float16)p1;
    }
    v16h ap = load_a_frag(s_p, 32, 0);     // re-fragment P as A (K=32) via LDS
    o0 = wmma32(ap, load_b_fragT(VTb, T_, 0,  j0), o0);
    o1 = wmma32(ap, load_b_fragT(VTb, T_, 16, j0), o1);
    o2 = wmma32(ap, load_b_fragT(VTb, T_, 32, j0), o2);
    o3 = wmma32(ap, load_b_fragT(VTb, T_, 48, j0), o3);
  }
#pragma unroll
  for (int i=0;i<8;++i){
    float inv = 1.f/lr[i];
    int M = i + 8*hi;
    size_t base = ((size_t)(b*T_ + qt*16 + M))*D_ + hh*HD_ + n;
    o16[base]    = (_Float16)(o0[i]*inv);
    o16[base+16] = (_Float16)(o1[i]*inv);
    o16[base+32] = (_Float16)(o2[i]*inv);
    o16[base+48] = (_Float16)(o3[i]*inv);
  }
}

extern "C" void kernel_launch(void* const* d_in, const int* in_sizes, int n_in,
                              void* d_out, int out_size, void* d_ws, size_t ws_size,
                              hipStream_t stream)
{
  const float* x        = (const float*)d_in[0];
  const int*   t        = (const int*)  d_in[1];
  const float* tsg      = (const float*)d_in[2];
  const float* mask     = (const float*)d_in[3];
  const float* noise    = (const float*)d_in[4];
  const float* alpha_bar= (const float*)d_in[6];
  const float* W_enc    = (const float*)d_in[7];
  const float* b_enc    = (const float*)d_in[8];
  const float* te_freqs = (const float*)d_in[9];
  const float* W_temb   = (const float*)d_in[10];
  const float* b_temb   = (const float*)d_in[11];
  const float* th_a     = (const float*)d_in[12];
  const float* th_b     = (const float*)d_in[13];
  const float* th_freqs = (const float*)d_in[14];
  const float* W_film   = (const float*)d_in[15];
  const float* b_film   = (const float*)d_in[16];
  const float* gru_Wih  = (const float*)d_in[17];
  const float* gru_Whh  = (const float*)d_in[18];
  const float* gru_bih  = (const float*)d_in[19];
  const float* gru_bhh  = (const float*)d_in[20];
  const float* ode_W1   = (const float*)d_in[21];
  const float* ode_b1   = (const float*)d_in[22];
  const float* ode_W2   = (const float*)d_in[23];
  const float* ode_b2   = (const float*)d_in[24];
  const float* tr_Wqkv  = (const float*)d_in[25];
  const float* tr_bqkv  = (const float*)d_in[26];
  const float* tr_Wo    = (const float*)d_in[27];
  const float* tr_bo    = (const float*)d_in[28];
  const float* ln1w     = (const float*)d_in[29];
  const float* ln1b     = (const float*)d_in[30];
  const float* ln2w     = (const float*)d_in[31];
  const float* ln2b     = (const float*)d_in[32];
  const float* tr_W1    = (const float*)d_in[33];
  const float* tr_b1    = (const float*)d_in[34];
  const float* tr_W2    = (const float*)d_in[35];
  const float* tr_b2    = (const float*)d_in[36];
  const float* trn_w    = (const float*)d_in[37];
  const float* trn_b    = (const float*)d_in[38];
  (void)in_sizes; (void)n_in; (void)out_size; (void)ws_size;

  char* ws = (char*)d_ws;
  size_t off = 0;
  auto take = [&](size_t bytes)->char*{
    char* p = ws + off; off = (off + bytes + 255) & ~(size_t)255; return p;
  };
  _Float16* wih16  = (_Float16*)take((size_t)256*768*2);
  _Float16* whh16  = (_Float16*)take((size_t)256*768*2);
  _Float16* ow1_16 = (_Float16*)take((size_t)256*256*2);
  _Float16* ow2_16 = (_Float16*)take((size_t)256*256*2);
  _Float16* wqkv16 = (_Float16*)take((size_t)4*256*768*2);
  _Float16* wo16   = (_Float16*)take((size_t)4*256*256*2);
  _Float16* w1_16  = (_Float16*)take((size_t)4*256*1024*2);
  _Float16* w2_16  = (_Float16*)take((size_t)4*1024*256*2);
  float*    te_buf = (float*)   take((size_t)B_*D_*4);
  float*    gb_buf = (float*)   take((size_t)B_*2*D_*4);
  _Float16* h16    = (_Float16*)take((size_t)BT_*D_*2);
  float*    Hseq   = (float*)   take((size_t)BT_*D_*4);
  float*    Ht     = (float*)   take((size_t)BT_*D_*4);
  _Float16* u16    = (_Float16*)take((size_t)BT_*D_*2);
  _Float16* qkv16  = (_Float16*)take((size_t)BT_*768*2);
  _Float16* o16    = (_Float16*)take((size_t)BT_*D_*2);
  _Float16* mid16  = (_Float16*)take((size_t)BT_*1024*2);
  _Float16* VT     = (_Float16*)take((size_t)B_*H_*HD_*T_*2);

  auto cvtf = [&](const float* s, _Float16* dd, int K, int N){
    int total = (K/32)*(N/16)*32;
    k_cvt_frag<<<(total+255)/256, 256, 0, stream>>>(s, dd, K, N);
  };
  cvtf(gru_Wih, wih16, 256, 768);
  cvtf(gru_Whh, whh16, 256, 768);
  cvtf(ode_W1,  ow1_16, 256, 256);
  cvtf(ode_W2,  ow2_16, 256, 256);
  for (int l=0;l<4;++l){
    cvtf(tr_Wqkv + (size_t)l*256*768,  wqkv16 + (size_t)l*256*768,  256, 768);
    cvtf(tr_Wo   + (size_t)l*256*256,  wo16   + (size_t)l*256*256,  256, 256);
    cvtf(tr_W1   + (size_t)l*256*1024, w1_16  + (size_t)l*256*1024, 256, 1024);
    cvtf(tr_W2   + (size_t)l*1024*256, w2_16  + (size_t)l*1024*256, 1024, 256);
  }

  k_te_film<<<1,256,0,stream>>>(t, te_freqs, W_temb, b_temb, W_film, b_film, te_buf, gb_buf);
  k_embed<<<BT_,256,0,stream>>>(x, t, tsg, mask, noise, alpha_bar, W_enc, b_enc,
                                th_a, th_b, th_freqs, te_buf, gb_buf, h16);
  k_scan<<<1,256,0,stream>>>(h16, tsg, wih16, whh16, gru_bih, gru_bhh,
                             ow1_16, ode_b1, ow2_16, ode_b2, Hseq);
  hipMemcpyAsync(Ht, Hseq, (size_t)BT_*D_*4, hipMemcpyDeviceToDevice, stream);

  for (int l=0; l<4; ++l){
    k_ln<<<BT_/8,256,0,stream>>>(Ht, ln1w + l*256, ln1b + l*256, u16, nullptr, nullptr);
    { int M=BT_, N=768, K=256;
      k_gemm<<<(M/16)*(N/256),256,16*K*2,stream>>>(u16, wqkv16 + (size_t)l*256*768, tr_bqkv + l*768,
                                                   nullptr, nullptr, qkv16, M,N,K, 0); }
    k_vt<<<BT_,256,0,stream>>>(qkv16, VT);
    k_attn<<<B_*H_*(T_/16),32,0,stream>>>(qkv16, VT, o16);
    { int M=BT_, N=256, K=256;
      k_gemm<<<(M/16)*(N/256),256,16*K*2,stream>>>(o16, wo16 + (size_t)l*256*256, tr_bo + l*256,
                                                   Ht, Ht, nullptr, M,N,K, 0); }
    k_ln<<<BT_/8,256,0,stream>>>(Ht, ln2w + l*256, ln2b + l*256, u16, nullptr, nullptr);
    { int M=BT_, N=1024, K=256;
      k_gemm<<<(M/16)*(N/256),256,16*K*2,stream>>>(u16, w1_16 + (size_t)l*256*1024, tr_b1 + l*1024,
                                                   nullptr, nullptr, mid16, M,N,K, 1); }
    { int M=BT_, N=256, K=1024;
      k_gemm<<<(M/16)*(N/256),256,16*K*2,stream>>>(mid16, w2_16 + (size_t)l*1024*256, tr_b2 + l*256,
                                                   Ht, Ht, nullptr, M,N,K, 0); }
  }
  // final LN fused with Hseq + Ht add, straight into d_out
  k_ln<<<BT_/8,256,0,stream>>>(Ht, trn_w, trn_b, nullptr, (float*)d_out, Hseq);
}
